// PlasticDualAttentionBlock_29850022707347
// MI455X (gfx1250) — matrix operand order repaired and checked by reading, not compiled, not run on hardware
//
#include <hip/hip_runtime.h>
#include <hip/hip_bf16.h>

// ---------------------------------------------------------------------------
// Problem constants: B=4, T=2048, C=1024, H=16, D=64, FF=4096
// ---------------------------------------------------------------------------
constexpr int BATCH = 4;
constexpr int SEQ   = 2048;
constexpr int EMB   = 1024;
constexpr int NH    = 16;
constexpr int HD    = 64;
constexpr int FFD   = 4096;
constexpr int QKV_LD = 3 * EMB;                 // 3072
constexpr size_t ROWS = (size_t)BATCH * SEQ;    // 8192

typedef __attribute__((ext_vector_type(16))) __bf16 v16bf;
typedef __attribute__((ext_vector_type(8)))  float  v8f;
typedef int v4i_vs __attribute__((vector_size(16)));   // matches builtin param type

union FragU { uint4 u[2]; v16bf v; };

// ---------------------------------------------------------------------------
// gfx1250 async memory->LDS copy (ASYNCcnt-tracked), with sync fallback.
// ---------------------------------------------------------------------------
#if __has_builtin(__builtin_amdgcn_global_load_async_to_lds_b128)
#define HAVE_ASYNC_LDS 1
#else
#define HAVE_ASYNC_LDS 0
#endif

__device__ __forceinline__ void copy16_to_lds(const void* g, void* l) {
#if HAVE_ASYNC_LDS
    __builtin_amdgcn_global_load_async_to_lds_b128(
        (__attribute__((address_space(1))) v4i_vs*)g,
        (__attribute__((address_space(3))) v4i_vs*)l, 0, 0);
#else
    *(uint4*)l = *(const uint4*)g;
#endif
}

template <int N>
__device__ __forceinline__ void wait_async() {
#if HAVE_ASYNC_LDS && __has_builtin(__builtin_amdgcn_s_wait_asynccnt)
    __builtin_amdgcn_s_wait_asynccnt(N);   // immediate required
#endif
}

__device__ __forceinline__ float gelu_f(float x) {
    return 0.5f * x * (1.0f + erff(x * 0.70710678118654752f));
}

// ---------------------------------------------------------------------------
// fp32 -> bf16 conversion (straight), and transposing variant for weights:
// in[K,N] (row major) -> out[N,K] so GEMM B tiles are K-contiguous per row.
// ---------------------------------------------------------------------------
__global__ void cvt_bf16_kernel(const float* __restrict__ s, __bf16* __restrict__ d, int n) {
    int i = blockIdx.x * blockDim.x + threadIdx.x;
    if (i < n) d[i] = (__bf16)s[i];
}

__global__ void cvt_bf16_t_kernel(const float* __restrict__ s, __bf16* __restrict__ d,
                                  int K, int N) {
    int i = blockIdx.x * blockDim.x + threadIdx.x;
    if (i < K * N) {
        const int k = i / N, n = i - k * N;
        d[(size_t)n * K + k] = (__bf16)s[i];
    }
}

// ---------------------------------------------------------------------------
// bf16 WMMA GEMM:  C[M,N] = A[M,K] * Bt[N,K]^T + bias
// 256 threads = 8 waves (4 M x 2 N); block tile 64x128; K-step 32.
// Double-buffered async-to-LDS staging; all fragment reads issued before the
// WMMA batch so one s_wait_dscnt covers them and the 4 WMMAs go back-to-back.
// mode 0: f32 out; mode 1: bf16 out; mode 2: gelu -> bf16 out
// ---------------------------------------------------------------------------
#define GEMM_BM 64
#define GEMM_BN 128
#define GEMM_BK 32

__global__ __launch_bounds__(256)
void gemm_bf16_kernel(const __bf16* __restrict__ A, const __bf16* __restrict__ Bt,
                      const float* __restrict__ bias,
                      float* __restrict__ Cf, __bf16* __restrict__ Cb,
                      int M, int N, int K, int mode) {
    __shared__ __align__(16) __bf16 As[2][GEMM_BM][GEMM_BK];   // 8 KB
    __shared__ __align__(16) __bf16 Bs[2][GEMM_BN][GEMM_BK];   // 16 KB

    const int tid  = threadIdx.x;
    const int lane = tid & 31;
    const int wave = tid >> 5;
    const int wm = wave & 3;            // 16-row sub-tile
    const int wn = wave >> 2;           // 64-col sub-tile
    const int bm = blockIdx.y * GEMM_BM;
    const int bn = blockIdx.x * GEMM_BN;

    // staging: A 64x32 (1 chunk/thread), B 128x32 (2 chunks/thread)
    const int a_row = tid >> 2, a_seg = tid & 3;     // 64 rows x 4 x 8elem
    const int b_row = tid >> 1, b_half = tid & 1;    // 128 rows x 2 x 16elem
    const __bf16* Ag = A  + (size_t)(bm + a_row) * K + a_seg * 8;
    const __bf16* Bg = Bt + (size_t)(bn + b_row) * K + b_half * 16;

    v8f acc[4] = {{}, {}, {}, {}};
    const int m = lane & 15, g = lane >> 4, nin = lane & 15;

    // prologue: stage buffer 0
    copy16_to_lds(Ag,     &As[0][a_row][a_seg * 8]);
    copy16_to_lds(Bg,     &Bs[0][b_row][b_half * 16]);
    copy16_to_lds(Bg + 8, &Bs[0][b_row][b_half * 16 + 8]);

    const int nk = K / GEMM_BK;
    for (int i = 0; i < nk; ++i) {
        const int cur = i & 1;
        __syncthreads();                       // prev compute done -> other buf free
        const bool more = (i + 1) < nk;
        if (more) {
            const int k0 = (i + 1) * GEMM_BK;
            copy16_to_lds(Ag + k0,     &As[cur ^ 1][a_row][a_seg * 8]);
            copy16_to_lds(Bg + k0,     &Bs[cur ^ 1][b_row][b_half * 16]);
            copy16_to_lds(Bg + k0 + 8, &Bs[cur ^ 1][b_row][b_half * 16 + 8]);
            __builtin_prefetch(Ag + k0 + GEMM_BK, 0, 1);
            __builtin_prefetch(Bg + k0 + GEMM_BK, 0, 1);
            wait_async<3>();                   // cur buffer's data has landed
        } else {
            wait_async<0>();
        }
        __syncthreads();                       // ... in every wave

        // issue ALL fragment loads first, then the WMMA batch
        FragU af, bf[4];
        const __bf16* ap = &As[cur][wm * 16 + m][0];
        af.u[0] = *(const uint4*)(ap + g * 8);
        af.u[1] = *(const uint4*)(ap + 16 + g * 8);
        const __bf16* bp = &Bs[cur][wn * 64 + nin][0];   // row stride 32 elems
#pragma unroll
        for (int t = 0; t < 4; ++t) {
            bf[t].u[0] = *(const uint4*)(bp + t * (16 * GEMM_BK) + g * 8);
            bf[t].u[1] = *(const uint4*)(bp + t * (16 * GEMM_BK) + 16 + g * 8);
        }
#pragma unroll
        for (int t = 0; t < 4; ++t) {
            acc[t] = __builtin_amdgcn_wmma_f32_16x16x32_bf16(
                false, af.v, false, bf[t].v, (short)0, acc[t], false, false);
        }
    }

    const int row0 = bm + wm * 16 + g * 8;
#pragma unroll
    for (int t = 0; t < 4; ++t) {
        const int col = bn + wn * 64 + t * 16 + nin;
        const float bia = bias[col];
#pragma unroll
        for (int r = 0; r < 8; ++r) {
            const float v = acc[t][r] + bia;
            const size_t idx = (size_t)(row0 + r) * N + col;
            if (mode == 0)      Cf[idx] = v;
            else if (mode == 1) Cb[idx] = (__bf16)v;
            else                Cb[idx] = (__bf16)gelu_f(v);
        }
    }
}

// ---------------------------------------------------------------------------
// Flash attention over packed qkv[8192, 3072] bf16.
// Grid (T/64, H, B); 128 threads = 4 waves; wave owns 16 query rows.
// K tile async-staged (row-contiguous), V tile staged transposed; fragment
// load batches precede each WMMA batch.
// ---------------------------------------------------------------------------
__global__ __launch_bounds__(128)
void attn_kernel(const __bf16* __restrict__ qkv, __bf16* __restrict__ out) {
    const int qt = blockIdx.x, h = blockIdx.y, b = blockIdx.z;
    const int wave = threadIdx.x >> 5, lane = threadIdx.x & 31;
    const int g = lane >> 4, ln = lane & 15;
    const size_t brow = (size_t)b * SEQ;
    const int q0 = qt * 64 + wave * 16;
    const int tid = threadIdx.x;

    __shared__ __align__(16) __bf16 Ks[32][64];        // [key][d]    4 KB
    __shared__ __align__(16) __bf16 Vt[64][32];        // [d][key]    4 KB
    __shared__ __align__(16) __bf16 Plds[4][16][32];   //             4 KB

    // Q fragments (two 32-wide d-halves), contiguous b128 global loads
    FragU qf0, qf1;
    {
        const __bf16* qp = qkv + (brow + q0 + ln) * QKV_LD + h * HD;
        qf0.u[0] = *(const uint4*)(qp + g * 8);
        qf0.u[1] = *(const uint4*)(qp + 16 + g * 8);
        qf1.u[0] = *(const uint4*)(qp + 32 + g * 8);
        qf1.u[1] = *(const uint4*)(qp + 48 + g * 8);
    }

    v8f o0 = {}, o1 = {}, o2 = {}, o3 = {};
    float mrow[8], lrow[8];
#pragma unroll
    for (int r = 0; r < 8; ++r) { mrow[r] = -3.0e38f; lrow[r] = 0.0f; }
    const float scale = 0.125f;   // 1/sqrt(64)

    // staging indices
    const int k_key = tid >> 2, k_seg = tid & 3;       // K: 32 keys x 4 x 8elem

    for (int j0 = 0; j0 < SEQ; j0 += 32) {
        __syncthreads();   // previous chunk fully consumed
        // --- stage K tile (async, layout-preserving) ---
        copy16_to_lds(qkv + (brow + j0 + k_key) * QKV_LD + EMB + h * HD + k_seg * 8,
                      &Ks[k_key][k_seg * 8]);
        // --- stage V tile transposed (sync scatter) ---
#pragma unroll
        for (int c = 0; c < 2; ++c) {
            const int chunk = tid * 2 + c;             // 0..255
            const int vk = chunk >> 3, vseg = chunk & 7;
            const __bf16* vg = qkv + (brow + j0 + vk) * QKV_LD + 2 * EMB + h * HD + vseg * 8;
            FragU tmp; tmp.u[0] = *(const uint4*)vg;
#pragma unroll
            for (int e = 0; e < 8; ++e) Vt[vseg * 8 + e][vk] = tmp.v[e];
        }
        wait_async<0>();
        __syncthreads();

        // --- S = Q K^T: two 16-key tiles, d = 64 in two 32-steps ---
        v8f s0 = {}, s1 = {};
        {
            FragU kf[4];
            kf[0].u[0] = *(const uint4*)&Ks[ln][g * 8];
            kf[0].u[1] = *(const uint4*)&Ks[ln][16 + g * 8];
            kf[1].u[0] = *(const uint4*)&Ks[ln][32 + g * 8];
            kf[1].u[1] = *(const uint4*)&Ks[ln][48 + g * 8];
            kf[2].u[0] = *(const uint4*)&Ks[16 + ln][g * 8];
            kf[2].u[1] = *(const uint4*)&Ks[16 + ln][16 + g * 8];
            kf[3].u[0] = *(const uint4*)&Ks[16 + ln][32 + g * 8];
            kf[3].u[1] = *(const uint4*)&Ks[16 + ln][48 + g * 8];
            s0 = __builtin_amdgcn_wmma_f32_16x16x32_bf16(false, qf0.v, false, kf[0].v, (short)0, s0, false, false);
            s0 = __builtin_amdgcn_wmma_f32_16x16x32_bf16(false, qf1.v, false, kf[1].v, (short)0, s0, false, false);
            s1 = __builtin_amdgcn_wmma_f32_16x16x32_bf16(false, qf0.v, false, kf[2].v, (short)0, s1, false, false);
            s1 = __builtin_amdgcn_wmma_f32_16x16x32_bf16(false, qf1.v, false, kf[3].v, (short)0, s1, false, false);
        }

        // --- online softmax (rows live in VGPR idx + lane half) ---
        float mnew[8], alpha[8];
#pragma unroll
        for (int r = 0; r < 8; ++r) {
            s0[r] *= scale; s1[r] *= scale;
            float t = fmaxf(s0[r], s1[r]);
#pragma unroll
            for (int off = 1; off < 16; off <<= 1) t = fmaxf(t, __shfl_xor(t, off, 32));
            mnew[r]  = fmaxf(mrow[r], t);
            alpha[r] = __expf(mrow[r] - mnew[r]);
            mrow[r]  = mnew[r];
        }
        v8f p0, p1;
#pragma unroll
        for (int r = 0; r < 8; ++r) {
            p0[r] = __expf(s0[r] - mnew[r]);
            p1[r] = __expf(s1[r] - mnew[r]);
            float t = p0[r] + p1[r];
#pragma unroll
            for (int off = 1; off < 16; off <<= 1) t += __shfl_xor(t, off, 32);
            lrow[r] = lrow[r] * alpha[r] + t;
            o0[r] *= alpha[r]; o1[r] *= alpha[r]; o2[r] *= alpha[r]; o3[r] *= alpha[r];
        }

        // --- restage P (16x32): C-layout -> A-layout via LDS ---
        __syncthreads();
#pragma unroll
        for (int r = 0; r < 8; ++r) {
            Plds[wave][g * 8 + r][ln]      = (__bf16)p0[r];
            Plds[wave][g * 8 + r][16 + ln] = (__bf16)p1[r];
        }
        __syncthreads();

        // --- O += P V : batch all fragment loads, then 4 WMMAs ---
        FragU pf, vf[4];
        {
            const __bf16* pp = &Plds[wave][ln][0];
            pf.u[0] = *(const uint4*)(pp + g * 8);
            pf.u[1] = *(const uint4*)(pp + 16 + g * 8);
        }
#pragma unroll
        for (int t = 0; t < 4; ++t) {
            vf[t].u[0] = *(const uint4*)&Vt[t * 16 + ln][g * 8];
            vf[t].u[1] = *(const uint4*)&Vt[t * 16 + ln][16 + g * 8];
        }
        o0 = __builtin_amdgcn_wmma_f32_16x16x32_bf16(false, pf.v, false, vf[0].v, (short)0, o0, false, false);
        o1 = __builtin_amdgcn_wmma_f32_16x16x32_bf16(false, pf.v, false, vf[1].v, (short)0, o1, false, false);
        o2 = __builtin_amdgcn_wmma_f32_16x16x32_bf16(false, pf.v, false, vf[2].v, (short)0, o2, false, false);
        o3 = __builtin_amdgcn_wmma_f32_16x16x32_bf16(false, pf.v, false, vf[3].v, (short)0, o3, false, false);
    }

    // --- normalize + store bf16 ---
#pragma unroll
    for (int r = 0; r < 8; ++r) {
        const float inv = 1.0f / lrow[r];
        __bf16* op = out + (brow + q0 + g * 8 + r) * EMB + h * HD;
        op[ln]      = (__bf16)(o0[r] * inv);
        op[16 + ln] = (__bf16)(o1[r] * inv);
        op[32 + ln] = (__bf16)(o2[r] * inv);
        op[48 + ln] = (__bf16)(o3[r] * inv);
    }
}

// ---------------------------------------------------------------------------
// y = LayerNorm(a + res) * gamma + beta ; f32 out (+ optional bf16 copy)
// ---------------------------------------------------------------------------
__global__ __launch_bounds__(256)
void add_ln_kernel(const float* __restrict__ a, const float* __restrict__ res,
                   const float* __restrict__ gamma, const float* __restrict__ beta,
                   float* __restrict__ outf, __bf16* __restrict__ outb) {
    const int row = blockIdx.x;
    const int tid = threadIdx.x;
    const float4 av = ((const float4*)(a   + (size_t)row * EMB))[tid];
    const float4 rv = ((const float4*)(res + (size_t)row * EMB))[tid];
    float v[4] = { av.x + rv.x, av.y + rv.y, av.z + rv.z, av.w + rv.w };
    float s  = v[0] + v[1] + v[2] + v[3];
    float ss = v[0]*v[0] + v[1]*v[1] + v[2]*v[2] + v[3]*v[3];
#pragma unroll
    for (int off = 16; off >= 1; off >>= 1) {
        s  += __shfl_xor(s,  off, 32);
        ss += __shfl_xor(ss, off, 32);
    }
    __shared__ float red[2][8];
    const int wave = tid >> 5, lane = tid & 31;
    if (lane == 0) { red[0][wave] = s; red[1][wave] = ss; }
    __syncthreads();
    float tot = 0.0f, tot2 = 0.0f;
#pragma unroll
    for (int w = 0; w < 8; ++w) { tot += red[0][w]; tot2 += red[1][w]; }
    const float mu   = tot * (1.0f / EMB);
    const float var  = tot2 * (1.0f / EMB) - mu * mu;
    const float rstd = rsqrtf(var + 1e-5f);
#pragma unroll
    for (int i = 0; i < 4; ++i) {
        const int c = tid * 4 + i;
        const float y = (v[i] - mu) * rstd * gamma[c] + beta[c];
        outf[(size_t)row * EMB + c] = y;
        if (outb) outb[(size_t)row * EMB + c] = (__bf16)y;
    }
}

// ---------------------------------------------------------------------------
// Host orchestration
// ---------------------------------------------------------------------------
extern "C" void kernel_launch(void* const* d_in, const int* in_sizes, int n_in,
                              void* d_out, int out_size, void* d_ws, size_t ws_size,
                              hipStream_t stream) {
    const float* x     = (const float*)d_in[0];
    const float* W_qkv = (const float*)d_in[1];
    const float* b_qkv = (const float*)d_in[2];
    const float* W_out = (const float*)d_in[3];
    const float* b_out = (const float*)d_in[4];
    const float* W_ff1 = (const float*)d_in[5];
    const float* b_ff1 = (const float*)d_in[6];
    const float* W_ff2 = (const float*)d_in[7];
    const float* b_ff2 = (const float*)d_in[8];
    const float* g1    = (const float*)d_in[9];
    const float* beta1 = (const float*)d_in[10];
    const float* g2    = (const float*)d_in[11];
    const float* beta2 = (const float*)d_in[12];
    float* out = (float*)d_out;

    // workspace carve (~280 MiB)
    char* p = (char*)d_ws;
    auto carve = [&](size_t bytes) -> void* {
        void* r = (void*)p;
        p += (bytes + 255) & ~(size_t)255;
        return r;
    };
    __bf16* xb   = (__bf16*)carve(ROWS * EMB * 2);
    __bf16* wqkv = (__bf16*)carve((size_t)EMB * 3 * EMB * 2);   // transposed [3C][C]
    __bf16* wout = (__bf16*)carve((size_t)EMB * EMB * 2);       // transposed [C][C]
    __bf16* wff1 = (__bf16*)carve((size_t)EMB * FFD * 2);       // transposed [FF][C]
    __bf16* wff2 = (__bf16*)carve((size_t)FFD * EMB * 2);       // transposed [C][FF]
    __bf16* qkvb = (__bf16*)carve(ROWS * 3 * EMB * 2);
    __bf16* attb = (__bf16*)carve(ROWS * EMB * 2);
    float*  aprj = (float*) carve(ROWS * EMB * 4);
    float*  hf   = (float*) carve(ROWS * EMB * 4);
    __bf16* hb   = (__bf16*)carve(ROWS * EMB * 2);
    __bf16* ff1b = (__bf16*)carve(ROWS * FFD * 2);
    float*  ff2f = (float*) carve(ROWS * EMB * 4);

    auto cvt = [&](const float* s, __bf16* d, size_t n) {
        cvt_bf16_kernel<<<dim3((unsigned)((n + 255) / 256)), dim3(256), 0, stream>>>(s, d, (int)n);
    };
    auto cvtT = [&](const float* s, __bf16* d, int K, int N) {
        size_t n = (size_t)K * N;
        cvt_bf16_t_kernel<<<dim3((unsigned)((n + 255) / 256)), dim3(256), 0, stream>>>(s, d, K, N);
    };
    cvt (x,     xb,   ROWS * EMB);
    cvtT(W_qkv, wqkv, EMB, 3 * EMB);
    cvtT(W_out, wout, EMB, EMB);
    cvtT(W_ff1, wff1, EMB, FFD);
    cvtT(W_ff2, wff2, FFD, EMB);

    const dim3 blk256(256);
    // QKV projection -> bf16 packed qkv
    gemm_bf16_kernel<<<dim3(3 * EMB / GEMM_BN, (unsigned)(ROWS / GEMM_BM)), blk256, 0, stream>>>(
        xb, wqkv, b_qkv, nullptr, qkvb, (int)ROWS, 3 * EMB, EMB, 1);
    // flash attention -> bf16
    attn_kernel<<<dim3(SEQ / 64, NH, BATCH), dim3(128), 0, stream>>>(qkvb, attb);
    // output projection -> f32
    gemm_bf16_kernel<<<dim3(EMB / GEMM_BN, (unsigned)(ROWS / GEMM_BM)), blk256, 0, stream>>>(
        attb, wout, b_out, aprj, nullptr, (int)ROWS, EMB, EMB, 0);
    // h = LN(attn_proj + x)
    add_ln_kernel<<<dim3((unsigned)ROWS), blk256, 0, stream>>>(aprj, x, g1, beta1, hf, hb);
    // FF1 + GELU -> bf16
    gemm_bf16_kernel<<<dim3(FFD / GEMM_BN, (unsigned)(ROWS / GEMM_BM)), blk256, 0, stream>>>(
        hb, wff1, b_ff1, nullptr, ff1b, (int)ROWS, FFD, EMB, 2);
    // FF2 -> f32
    gemm_bf16_kernel<<<dim3(EMB / GEMM_BN, (unsigned)(ROWS / GEMM_BM)), blk256, 0, stream>>>(
        ff1b, wff2, b_ff2, ff2f, nullptr, (int)ROWS, EMB, FFD, 0);
    // out = LN(ff2 + h)
    add_ln_kernel<<<dim3((unsigned)ROWS), blk256, 0, stream>>>(ff2f, hf, g2, beta2, out, nullptr);
}
